// MultiHeadCrossAttention_2869038154330
// MI455X (gfx1250) — matrix operand order repaired and checked by reading, not compiled, not run on hardware
//
#include <hip/hip_runtime.h>
#include <hip/hip_bf16.h>
#include <math.h>

typedef _Float16 half_t;
typedef __attribute__((ext_vector_type(16))) _Float16 v16h;
typedef __attribute__((ext_vector_type(8)))  _Float16 h8;
typedef __attribute__((ext_vector_type(8)))  float    v8f;
typedef __attribute__((ext_vector_type(4)))  float    f4;
typedef __attribute__((ext_vector_type(4)))  int      i4;

#define D_MODEL   1024
#define NUM_HEADS 16
#define HEAD_DIM  64
#define BATCH     2
#define SEQ       2048

// ---------------------------------------------------------------------------
// CDNA5 async global->LDS copies (ASYNCcnt), guarded so compile never breaks.
// Probe-confirmed prototype: (int4 as(1)* src, int4 as(3)* dst, imm, imm cpol)
// ---------------------------------------------------------------------------
#if defined(__has_builtin)
#if __has_builtin(__builtin_amdgcn_global_load_async_to_lds_b128) && \
    __has_builtin(__builtin_amdgcn_s_wait_asynccnt)
#define HAS_ASYNC 1
#endif
#endif
#ifndef HAS_ASYNC
#define HAS_ASYNC 0
#endif

typedef __attribute__((address_space(1))) i4 gi4;
typedef __attribute__((address_space(3))) i4 li4;

__device__ __forceinline__ void async_copy16(const half_t* g, half_t* l) {
#if HAS_ASYNC
    __builtin_amdgcn_global_load_async_to_lds_b128((gi4*)g, (li4*)l, 0, 0);
#else
    *(h8*)l = *(const h8*)g;
#endif
}
__device__ __forceinline__ void async_wait0() {
#if HAS_ASYNC
    __builtin_amdgcn_s_wait_asynccnt(0);
#endif
}
__device__ __forceinline__ void async_wait3() {
#if HAS_ASYNC
    __builtin_amdgcn_s_wait_asynccnt(3);
#endif
}

// ---------------------------------------------------------------------------
// WMMA fragment helpers (layouts per CDNA5 ISA 7.12.2, wave32)
// A 16x32 f16: row = lane%16, element i -> k = (i/8)*16 + (lane/16)*8 + (i%8)
// B 32x16 f16: col n = lane%16, element i -> k = (i/8)*16 + (lane/16)*8 + (i%8)
// C/D 16x16 f32: col = lane%16, VGPR r -> row = r + 8*(lane/16)
// ---------------------------------------------------------------------------
__device__ __forceinline__ v16h load_a_frag(const half_t* rowp, int g) {
    h8 lo = *(const h8*)(rowp + g * 8);
    h8 hi = *(const h8*)(rowp + 16 + g * 8);
    v16h a;
#pragma unroll
    for (int i = 0; i < 8; ++i) { a[i] = lo[i]; a[i + 8] = hi[i]; }
    return a;
}
__device__ __forceinline__ v16h load_b_frag(const half_t* colp, int g) {
    h8 lo = *(const h8*)(colp + g * 8);
    h8 hi = *(const h8*)(colp + 16 + g * 8);
    v16h b;
#pragma unroll
    for (int i = 0; i < 8; ++i) { b[i] = lo[i]; b[i + 8] = hi[i]; }
    return b;
}
__device__ __forceinline__ v8f wmma_f16(v16h a, v16h b, v8f c) {
    return __builtin_amdgcn_wmma_f32_16x16x32_f16(false, a, false, b,
                                                  (short)0, c, false, false);
}
__device__ __forceinline__ float redmax16(float v) {
#pragma unroll
    for (int m = 1; m < 16; m <<= 1) v = fmaxf(v, __shfl_xor(v, m, 32));
    return v;
}
__device__ __forceinline__ float redsum16(float v) {
#pragma unroll
    for (int m = 1; m < 16; m <<= 1) v += __shfl_xor(v, m, 32);
    return v;
}

// ---------------------------------------------------------------------------
// Pre-pass: f32 -> f16 convert (x, y), 8 elems / thread
// ---------------------------------------------------------------------------
__global__ __launch_bounds__(256) void convert_f16_kernel(
    const float* __restrict__ src, half_t* __restrict__ dst) {
    size_t i = ((size_t)blockIdx.x * 256 + threadIdx.x) * 8;
    f4 a = *(const f4*)(src + i);
    f4 b = *(const f4*)(src + i + 4);
    h8 o;
#pragma unroll
    for (int j = 0; j < 4; ++j) { o[j] = (half_t)a[j]; o[4 + j] = (half_t)b[j]; }
    *(h8*)(dst + i) = o;
}

// ---------------------------------------------------------------------------
// Pre-pass: W[k][n] f32 -> Wt[n][k] f16 (64x64 tiles through LDS)
// ---------------------------------------------------------------------------
__global__ __launch_bounds__(256) void transpose_f16_kernel(
    const float* __restrict__ W, half_t* __restrict__ Wt, int N) {
    __shared__ half_t lT[64 * 72];
    const int n0 = blockIdx.x * 64, k0 = blockIdx.y * 64;
    const int tid = threadIdx.x;
#pragma unroll
    for (int p = 0; p < 16; ++p) {
        int idx = p * 256 + tid;
        int kk = idx >> 6, nn = idx & 63;
        lT[nn * 72 + kk] = (half_t)W[(size_t)(k0 + kk) * N + n0 + nn];
    }
    __syncthreads();
#pragma unroll
    for (int p = 0; p < 2; ++p) {
        int c = p * 256 + tid;
        int nn = c >> 3, seg = c & 7;
        *(h8*)(Wt + (size_t)(n0 + nn) * D_MODEL + k0 + seg * 8) =
            *(const h8*)(&lT[nn * 72 + seg * 8]);
    }
}

// ---------------------------------------------------------------------------
// Tiled GEMM, both operands pre-converted f16:
//   C[M,N] = Af16[M,1024] @ Bt[N,1024]^T + bias
// Block 256 thr (8 waves, 4x2), tile 128(M)x64(N), K-step 32,
// double-buffered LDS filled by async global->LDS b128 copies.
// MODE 0: scatter f16 K/V halves; MODE 1: scatter f16 Q; MODE 2: f32 out.
// ---------------------------------------------------------------------------
#define GT_M 128
#define GT_N 64
#define GT_K 32
#define LDA  40   // 32 + 8 pad (halfs)
#define LDB  48   // 32 + 16 pad (halfs)

__device__ __forceinline__ void gemm_stage(const half_t* __restrict__ Af,
                                           const half_t* __restrict__ Bt,
                                           half_t* lA, half_t* lB, int tileM,
                                           int tileN, int kt, int tid) {
    const int seg = tid & 3;       // 16B chunk within 32-half row
    const int row = tid >> 2;      // 0..63
#pragma unroll
    for (int p = 0; p < 2; ++p) {
        int r = row + p * 64;      // A rows 0..127
        async_copy16(Af + (size_t)(tileM + r) * D_MODEL + kt + seg * 8,
                     &lA[r * LDA + seg * 8]);
    }
    async_copy16(Bt + (size_t)(tileN + row) * D_MODEL + kt + seg * 8,
                 &lB[row * LDB + seg * 8]);
}

template <int MODE>
__global__ __launch_bounds__(256) void gemm_kernel(
    const half_t* __restrict__ Af, const half_t* __restrict__ Bt,
    const float* __restrict__ bias, half_t* __restrict__ out_h0,
    half_t* __restrict__ out_h1, float* __restrict__ out_f, int N) {
    __shared__ half_t lA[2][GT_M * LDA];
    __shared__ half_t lB[2][GT_N * LDB];

    const int tid = threadIdx.x;
    const int lane = tid & 31;
    const int wid = tid >> 5;
    const int waveM = wid >> 1;
    const int waveN = wid & 1;
    const int g = lane >> 4;
    const int l16 = lane & 15;
    const int tileM = blockIdx.y * GT_M;
    const int tileN = blockIdx.x * GT_N;

    v8f acc[2][2];
#pragma unroll
    for (int i = 0; i < 2; ++i)
#pragma unroll
        for (int j = 0; j < 2; ++j)
#pragma unroll
            for (int r = 0; r < 8; ++r) acc[i][j][r] = 0.0f;

    gemm_stage(Af, Bt, lA[0], lB[0], tileM, tileN, 0, tid);

    const int NK = D_MODEL / GT_K;  // 32
    for (int t = 0; t < NK; ++t) {
        const int cur = t & 1;
        if (t + 1 < NK) {
            gemm_stage(Af, Bt, lA[cur ^ 1], lB[cur ^ 1], tileM, tileN,
                       (t + 1) * GT_K, tid);
            async_wait3();   // oldest 3 (current tile) complete
        } else {
            async_wait0();
        }
        __syncthreads();

        v16h af[2], bf[2];
#pragma unroll
        for (int tm = 0; tm < 2; ++tm)
            af[tm] = load_a_frag(&lA[cur][(waveM * 32 + tm * 16 + l16) * LDA], g);
#pragma unroll
        for (int tn = 0; tn < 2; ++tn)
            bf[tn] = load_b_frag(&lB[cur][(waveN * 32 + tn * 16 + l16) * LDB], g);
#pragma unroll
        for (int tm = 0; tm < 2; ++tm)
#pragma unroll
            for (int tn = 0; tn < 2; ++tn)
                acc[tm][tn] = wmma_f16(af[tm], bf[tn], acc[tm][tn]);
        __syncthreads();
    }

#pragma unroll
    for (int tm = 0; tm < 2; ++tm) {
#pragma unroll
        for (int tn = 0; tn < 2; ++tn) {
            int col = tileN + waveN * 32 + tn * 16 + l16;
            float bval = bias[col];
#pragma unroll
            for (int r = 0; r < 8; ++r) {
                int row = tileM + waveM * 32 + tm * 16 + r + 8 * g;
                float val = acc[tm][tn][r] + bval;
                int b = row >> 11;
                int s = row & (SEQ - 1);
                if constexpr (MODE == 0) {
                    int h = col >> 7;
                    int within = col & 127;
                    int d = within & 63;
                    half_t* dst = (within < 64) ? out_h0 : out_h1;
                    dst[(((size_t)(b * NUM_HEADS + h)) * SEQ + s) * HEAD_DIM + d] =
                        (half_t)val;
                } else if constexpr (MODE == 1) {
                    int h = col >> 6;
                    int d = col & 63;
                    out_h0[(((size_t)(b * NUM_HEADS + h)) * SEQ + s) * HEAD_DIM + d] =
                        (half_t)val;
                } else {
                    out_f[(size_t)row * N + col] = val;
                }
            }
        }
    }
}

// ---------------------------------------------------------------------------
// Flash attention: block = 64 q rows of one (b,h); 4 waves x 16 rows.
// K chunk staged via async global->LDS; V transposed synchronously (its
// global loads overlap the async K DMA). Online softmax, f16 output.
// ---------------------------------------------------------------------------
#define AT_KC 32
#define LKK 80
#define LVT 48
#define LP  40

__global__ __launch_bounds__(128) void attn_kernel(
    const half_t* __restrict__ qbuf, const half_t* __restrict__ kbuf,
    const half_t* __restrict__ vbuf, const float* __restrict__ mask,
    half_t* __restrict__ attn_out) {
    __shared__ half_t lK[AT_KC * LKK];
    __shared__ half_t lV[HEAD_DIM * LVT];
    __shared__ half_t lP[4 * 16 * LP];

    const int tid = threadIdx.x;
    const int lane = tid & 31;
    const int wid = tid >> 5;
    const int g = lane >> 4;
    const int l16 = lane & 15;
    const int bh = blockIdx.y;
    const int b = bh >> 4;
    const int h = bh & 15;
    const int qbase = blockIdx.x * 64;

    const int qrow = qbase + wid * 16 + l16;
    const half_t* qrowp = qbuf + ((size_t)bh * SEQ + qrow) * HEAD_DIM;
    v16h aq0 = load_a_frag(qrowp, g);
    v16h aq1 = load_a_frag(qrowp + 32, g);

    v8f acc_o[4];
#pragma unroll
    for (int t = 0; t < 4; ++t)
#pragma unroll
        for (int r = 0; r < 8; ++r) acc_o[t][r] = 0.0f;

    float m_run[8], l_run[8];
#pragma unroll
    for (int r = 0; r < 8; ++r) { m_run[r] = -__builtin_inff(); l_run[r] = 0.0f; }

    const half_t* kg = kbuf + (size_t)bh * SEQ * HEAD_DIM;
    const half_t* vg = vbuf + (size_t)bh * SEQ * HEAD_DIM;
    const float* mbase0 = mask + ((size_t)(b * SEQ + qbase + wid * 16)) * SEQ;
    half_t* lPw = &lP[wid * 16 * LP];

    for (int kc = 0; kc < SEQ; kc += AT_KC) {
        // async-stage K chunk (32x64 halfs = 256 x 16B)
#pragma unroll
        for (int p = 0; p < 2; ++p) {
            int idx = p * 128 + tid;
            int row = idx >> 3, seg = idx & 7;
            async_copy16(kg + ((size_t)(kc + row)) * HEAD_DIM + seg * 8,
                         &lK[row * LKK + seg * 8]);
        }
        // V chunk transposed into lV[d][kj] (synchronous)
#pragma unroll
        for (int p = 0; p < 2; ++p) {
            int idx = p * 128 + tid;
            int row = idx >> 3, seg = idx & 7;
            h8 v = *(const h8*)(vg + ((size_t)(kc + row)) * HEAD_DIM + seg * 8);
#pragma unroll
            for (int j = 0; j < 8; ++j) lV[(seg * 8 + j) * LVT + row] = v[j];
        }
        async_wait0();
        __syncthreads();

        v8f s[2];
#pragma unroll
        for (int tn = 0; tn < 2; ++tn) {
#pragma unroll
            for (int r = 0; r < 8; ++r) s[tn][r] = 0.0f;
            const half_t* bp = &lK[(tn * 16 + l16) * LKK];
            s[tn] = wmma_f16(aq0, load_b_frag(bp, g), s[tn]);
            s[tn] = wmma_f16(aq1, load_b_frag(bp + 32, g), s[tn]);
        }

        const float* mb = mbase0 + kc;
        float sv0[8], sv1[8], newmax[8];
#pragma unroll
        for (int r = 0; r < 8; ++r) {
            int rl = r + 8 * g;
            const float* mr = mb + (size_t)rl * SEQ;
            float a0 = s[0][r] * 0.125f + mr[l16];
            float a1 = s[1][r] * 0.125f + mr[16 + l16];
            sv0[r] = a0; sv1[r] = a1;
            float mx = redmax16(fmaxf(a0, a1));
            newmax[r] = fmaxf(m_run[r], mx);
        }
#pragma unroll
        for (int r = 0; r < 8; ++r) {
            float p0 = __expf(sv0[r] - newmax[r]);
            float p1 = __expf(sv1[r] - newmax[r]);
            float f = __expf(m_run[r] - newmax[r]);
            float rs = redsum16(p0 + p1);
            l_run[r] = l_run[r] * f + rs;
            m_run[r] = newmax[r];
#pragma unroll
            for (int t = 0; t < 4; ++t) acc_o[t][r] *= f;
            int rl = r + 8 * g;
            lPw[rl * LP + l16] = (half_t)p0;
            lPw[rl * LP + 16 + l16] = (half_t)p1;
        }

        v16h ap = load_a_frag(&lPw[l16 * LP], g);
#pragma unroll
        for (int t = 0; t < 4; ++t) {
            v16h bv = load_b_frag(&lV[(t * 16 + l16) * LVT], g);
            acc_o[t] = wmma_f16(ap, bv, acc_o[t]);
        }
        __syncthreads();
    }

    // normalize + write f16 [B][S][D_MODEL]
#pragma unroll
    for (int r = 0; r < 8; ++r) {
        int row = qbase + wid * 16 + r + 8 * g;
        float inv = 1.0f / l_run[r];
        half_t* op = attn_out + ((size_t)(b * SEQ + row)) * D_MODEL + h * HEAD_DIM;
#pragma unroll
        for (int t = 0; t < 4; ++t) op[t * 16 + l16] = (half_t)(acc_o[t][r] * inv);
    }
}

// ---------------------------------------------------------------------------
extern "C" void kernel_launch(void* const* d_in, const int* in_sizes, int n_in,
                              void* d_out, int out_size, void* d_ws,
                              size_t ws_size, hipStream_t stream) {
    const float* x    = (const float*)d_in[0];
    const float* y    = (const float*)d_in[1];
    const float* mask = (const float*)d_in[2];
    const float* Wkv  = (const float*)d_in[3];
    const float* bkv  = (const float*)d_in[4];
    const float* Wq   = (const float*)d_in[5];
    const float* bq   = (const float*)d_in[6];
    const float* Wo   = (const float*)d_in[7];
    const float* bo   = (const float*)d_in[8];
    float* out = (float*)d_out;

    const size_t MTOT = (size_t)BATCH * SEQ * D_MODEL;  // 4M
    half_t* xa   = (half_t*)d_ws;          // 4M halfs
    half_t* ya   = xa + MTOT;              // 4M
    half_t* wkvt = ya + MTOT;              // 2M (2048x1024)
    half_t* wqt  = wkvt + 2 * MTOT / 4;    // 1M
    half_t* wot  = wqt + MTOT / 4;         // 1M
    half_t* kb   = wot + MTOT / 4;         // 4M
    half_t* vb   = kb + MTOT;              // 4M
    half_t* qb   = vb + MTOT;              // 4M
    half_t* attn = qb + MTOT;              // 4M

    const int M = BATCH * SEQ;  // 4096
    dim3 blk(256);

    // pre-pass: f16 conversions and weight transposes
    convert_f16_kernel<<<dim3(MTOT / (256 * 8)), blk, 0, stream>>>(x, xa);
    convert_f16_kernel<<<dim3(MTOT / (256 * 8)), blk, 0, stream>>>(y, ya);
    transpose_f16_kernel<<<dim3(32, 16), blk, 0, stream>>>(Wkv, wkvt, 2 * D_MODEL);
    transpose_f16_kernel<<<dim3(16, 16), blk, 0, stream>>>(Wq, wqt, D_MODEL);
    transpose_f16_kernel<<<dim3(16, 16), blk, 0, stream>>>(Wo, wot, D_MODEL);

    gemm_kernel<0><<<dim3((2 * D_MODEL) / GT_N, M / GT_M), blk, 0, stream>>>(
        xa, wkvt, bkv, kb, vb, nullptr, 2 * D_MODEL);
    gemm_kernel<1><<<dim3(D_MODEL / GT_N, M / GT_M), blk, 0, stream>>>(
        ya, wqt, bq, qb, nullptr, nullptr, D_MODEL);
    attn_kernel<<<dim3(SEQ / 64, BATCH * NUM_HEADS), dim3(128), 0, stream>>>(
        qb, kb, vb, mask, attn);
    gemm_kernel<2><<<dim3(D_MODEL / GT_N, M / GT_M), blk, 0, stream>>>(
        attn, wot, bo, nullptr, nullptr, out, D_MODEL);
}
